// SAGE_Encoder_50276887167328
// MI455X (gfx1250) — matrix operand order repaired and checked by reading, not compiled, not run on hardware
//
#include <hip/hip_runtime.h>

typedef __attribute__((ext_vector_type(16))) __bf16 v16bf;
typedef __attribute__((ext_vector_type(8)))  float  v8f;

#define FEAT 128

// ---------------------------------------------------------------- utilities

__global__ void zero_f32_kernel(float* __restrict__ p, size_t n) {
    size_t i = (size_t)blockIdx.x * blockDim.x + threadIdx.x;
    size_t stride = (size_t)gridDim.x * blockDim.x;
    for (; i < n; i += stride) p[i] = 0.0f;
}

__global__ void count_deg_kernel(const int* __restrict__ esrc,
                                 const int* __restrict__ edst,
                                 int* __restrict__ deg_u,
                                 int* __restrict__ deg_m, int E) {
    int e = blockIdx.x * blockDim.x + threadIdx.x;
    if (e >= E) return;
    atomicAdd(deg_u + esrc[e], 1);
    atomicAdd(deg_m + edst[e], 1);
}

// ------------------------------------------------ weight pack: f32 -> bf16 B-fragments
// B (32x16, 16-bit) WMMA layout: lane = N (mod 16); lanes 0-15 hold
// K={0..7, 16..23}, lanes 16-31 hold K shifted by +8; VGPR j holds K = khalf + 2j + (j>=4?8:0), +1 (packed).
// Fragment array layout: frag[((kt*8 + nt)*32 + lane)*8 + j]  (uint32 = 2 bf16)

__device__ inline unsigned int pack2_bf16(float a, float b) {
    union { unsigned int u; __bf16 h[2]; } t;
    t.h[0] = (__bf16)a;
    t.h[1] = (__bf16)b;
    return t.u;
}

__global__ void pack_w_kernel(const float* __restrict__ W,
                              unsigned int* __restrict__ frag) {
    int kt = blockIdx.x >> 3;      // 0..3  (K tile of 32)
    int nt = blockIdx.x & 7;       // 0..7  (N tile of 16)
    int lane = threadIdx.x & 31;
    int n  = nt * 16 + (lane & 15);
    int kb = kt * 32 + ((lane >> 4) ? 8 : 0);
    unsigned int* dst = frag + ((size_t)((kt * 8 + nt) * 32 + lane) << 3);
#pragma unroll
    for (int j = 0; j < 8; ++j) {
        float a = W[(size_t)(kb + 2 * j + (j >= 4 ? 8 : 0)) * FEAT + n];
        float b = W[(size_t)(kb + 2 * j + (j >= 4 ? 8 : 0) + 1) * FEAT + n];
        dst[j] = pack2_bf16(a, b);
    }
}

// ------------------------------------------------ edge scatter (both directions)
// 32 threads per edge, 4 feature floats per thread; fp32 atomics land in L2
// (agg arrays ~61MB << 192MB L2).

__global__ void scatter_accum_kernel(const float* __restrict__ feat_u,
                                     const float* __restrict__ feat_m,
                                     const int* __restrict__ esrc,
                                     const int* __restrict__ edst,
                                     float* __restrict__ agg_m,
                                     float* __restrict__ agg_u, int E) {
    long long t = (long long)blockIdx.x * blockDim.x + threadIdx.x;
    long long e = t >> 5;
    if (e >= E) return;
    int q = (int)(t & 31) * 4;
    int s = esrc[e], d = edst[e];

    float4 vu = *reinterpret_cast<const float4*>(feat_u + (size_t)s * FEAT + q);
    float* pm = agg_m + (size_t)d * FEAT + q;
    unsafeAtomicAdd(pm + 0, vu.x);
    unsafeAtomicAdd(pm + 1, vu.y);
    unsafeAtomicAdd(pm + 2, vu.z);
    unsafeAtomicAdd(pm + 3, vu.w);

    float4 vm = *reinterpret_cast<const float4*>(feat_m + (size_t)d * FEAT + q);
    float* pu = agg_u + (size_t)s * FEAT + q;
    unsafeAtomicAdd(pu + 0, vm.x);
    unsafeAtomicAdd(pu + 1, vm.y);
    unsafeAtomicAdd(pu + 2, vm.z);
    unsafeAtomicAdd(pu + 3, vm.w);
}

// ------------------------------------------------ fused SAGE GEMM
// out = relu_res? self + relu(agg/deg @ Wl + b + self @ Wr)
//               : (agg/deg @ Wl + b + self @ Wr)
// A (16x32, 16-bit) WMMA layout: lane = M (mod 16); lanes 0-15 hold
// K={kb..kb+7, kb+16..kb+23}; lanes 16-31 hold K shifted by +8.

__device__ inline v16bf load_a_frag(const float* __restrict__ rowBase,
                                    int kb, int ahalf, float s) {
    const float4* p0 = reinterpret_cast<const float4*>(rowBase + kb + ahalf);
    const float4* p1 = reinterpret_cast<const float4*>(rowBase + kb + 16 + ahalf);
    float4 x0 = p0[0], x1 = p0[1], y0 = p1[0], y1 = p1[1];
    v16bf v;
    v[0]  = (__bf16)(x0.x * s); v[1]  = (__bf16)(x0.y * s);
    v[2]  = (__bf16)(x0.z * s); v[3]  = (__bf16)(x0.w * s);
    v[4]  = (__bf16)(x1.x * s); v[5]  = (__bf16)(x1.y * s);
    v[6]  = (__bf16)(x1.z * s); v[7]  = (__bf16)(x1.w * s);
    v[8]  = (__bf16)(y0.x * s); v[9]  = (__bf16)(y0.y * s);
    v[10] = (__bf16)(y0.z * s); v[11] = (__bf16)(y0.w * s);
    v[12] = (__bf16)(y1.x * s); v[13] = (__bf16)(y1.y * s);
    v[14] = (__bf16)(y1.z * s); v[15] = (__bf16)(y1.w * s);
    return v;
}

__device__ inline v16bf load_b_frag(const unsigned int* __restrict__ frag,
                                    int kt, int nt, int lane) {
    union { v16bf v; uint4 q[2]; } f;
    const uint4* p = reinterpret_cast<const uint4*>(
        frag + ((size_t)((kt * 8 + nt) * 32 + lane) << 3));
    f.q[0] = p[0];
    f.q[1] = p[1];
    return f.v;
}

template <bool RELU_RES>
__global__ void sage_gemm_kernel(const float* __restrict__ agg,
                                 const int* __restrict__ deg,
                                 const float* __restrict__ selfF,
                                 const unsigned int* __restrict__ wl,
                                 const unsigned int* __restrict__ wr,
                                 const float* __restrict__ bias,
                                 float* __restrict__ out, int M) {
    const int lane = threadIdx.x & 31;
    const int wave = threadIdx.x >> 5;      // 0..3, each wave does 2 N-tiles
    const int mt = blockIdx.x;

    int arow = mt * 16 + (lane & 15);
    if (arow >= M) arow = M - 1;            // clamp (M is a multiple of 16 here)
    const int ahalf = (lane >> 4) ? 8 : 0;
    const float* aggRow  = agg   + (size_t)arow * FEAT;
    const float* selfRow = selfF + (size_t)arow * FEAT;
    const int dg = deg[arow];
    const float scale = 1.0f / (float)(dg > 0 ? dg : 1);

    const int nt0 = wave * 2, nt1 = wave * 2 + 1;
    v8f c0 = {}; v8f c1 = {};

#pragma unroll
    for (int kt = 0; kt < 4; ++kt) {
        const int kb = kt * 32;
        v16bf aAgg  = load_a_frag(aggRow,  kb, ahalf, scale);
        v16bf aSelf = load_a_frag(selfRow, kb, ahalf, 1.0f);
        v16bf bl0 = load_b_frag(wl, kt, nt0, lane);
        v16bf br0 = load_b_frag(wr, kt, nt0, lane);
        v16bf bl1 = load_b_frag(wl, kt, nt1, lane);
        v16bf br1 = load_b_frag(wr, kt, nt1, lane);
        c0 = __builtin_amdgcn_wmma_f32_16x16x32_bf16(false, aAgg,  false, bl0,
                                                     (short)0, c0, false, false);
        c0 = __builtin_amdgcn_wmma_f32_16x16x32_bf16(false, aSelf, false, br0,
                                                     (short)0, c0, false, false);
        c1 = __builtin_amdgcn_wmma_f32_16x16x32_bf16(false, aAgg,  false, bl1,
                                                     (short)0, c1, false, false);
        c1 = __builtin_amdgcn_wmma_f32_16x16x32_bf16(false, aSelf, false, br1,
                                                     (short)0, c1, false, false);
    }

    // C/D layout: lanes 0-15 -> M = r, lanes 16-31 -> M = 8 + r; N = lane%16
    const int chalf = (lane >> 4) ? 8 : 0;
    const int cn = lane & 15;
    const int col0 = nt0 * 16 + cn;
    const int col1 = nt1 * 16 + cn;
    const float b0 = bias[col0];
    const float b1 = bias[col1];
#pragma unroll
    for (int r = 0; r < 8; ++r) {
        const int row = mt * 16 + chalf + r;
        if (row < M) {
            float v0 = c0[r] + b0;
            float v1 = c1[r] + b1;
            if (RELU_RES) {
                v0 = selfF[(size_t)row * FEAT + col0] + fmaxf(v0, 0.0f);
                v1 = selfF[(size_t)row * FEAT + col1] + fmaxf(v1, 0.0f);
            }
            out[(size_t)row * FEAT + col0] = v0;
            out[(size_t)row * FEAT + col1] = v1;
        }
    }
}

// ---------------------------------------------------------------- launcher

extern "C" void kernel_launch(void* const* d_in, const int* in_sizes, int n_in,
                              void* d_out, int out_size, void* d_ws, size_t ws_size,
                              hipStream_t stream) {
    (void)n_in; (void)out_size; (void)ws_size;

    const float* xu   = (const float*)d_in[0];
    const float* xm   = (const float*)d_in[1];
    const int*   esrc = (const int*)d_in[2];
    const int*   edst = (const int*)d_in[3];
    const float* W[8];
    for (int i = 0; i < 8; ++i) W[i] = (const float*)d_in[4 + i];
    // W order: Wl1_um, Wr1_um, Wl1_mu, Wr1_mu, Wl2_um, Wr2_um, Wl2_mu, Wr2_mu
    const float* b1_um = (const float*)d_in[12];
    const float* b1_mu = (const float*)d_in[13];
    const float* b2_um = (const float*)d_in[14];
    const float* b2_mu = (const float*)d_in[15];

    const int NU = in_sizes[0] / FEAT;
    const int NM = in_sizes[1] / FEAT;
    const int E  = in_sizes[2];

    float* out_u = (float*)d_out;
    float* out_m = (float*)d_out + (size_t)NU * FEAT;

    // workspace layout (contiguous: deg_u, deg_m, agg_u, agg_m, r_u, r_m, wfrag)
    char* ws = (char*)d_ws;
    size_t off = 0;
    auto alloc = [&](size_t bytes) -> char* {
        off = (off + 255) & ~(size_t)255;
        char* p = ws + off;
        off += bytes;
        return p;
    };
    int*   deg_u = (int*)alloc((size_t)NU * 4);
    int*   deg_m = (int*)alloc((size_t)NM * 4);
    float* agg_u = (float*)alloc((size_t)NU * FEAT * 4);
    float* agg_m = (float*)alloc((size_t)NM * FEAT * 4);
    float* agg_end = (float*)(ws + ((off + 255) & ~(size_t)255));
    float* r_u = (float*)alloc((size_t)NU * FEAT * 4);
    float* r_m = (float*)alloc((size_t)NM * FEAT * 4);
    unsigned int* wfrag = (unsigned int*)alloc((size_t)8 * 8192 * 4);

    const size_t zero_all_n = (size_t)(agg_end - (float*)deg_u); // deg+agg span
    const size_t zero_agg_n = (size_t)(agg_end - agg_u);         // agg span only

    const dim3 zgrid(4096), zblk(256);
    // 1. zero degrees + aggregation buffers
    zero_f32_kernel<<<zgrid, zblk, 0, stream>>>((float*)deg_u, zero_all_n);
    // 2. degree counts (shared by both layers)
    count_deg_kernel<<<(E + 255) / 256, 256, 0, stream>>>(esrc, edst, deg_u, deg_m, E);
    // 3. pack all 8 weight matrices into bf16 WMMA B-fragments
    for (int i = 0; i < 8; ++i)
        pack_w_kernel<<<32, 32, 0, stream>>>(W[i], wfrag + (size_t)i * 8192);

    const long long sthreads = (long long)E * 32;
    const int sblocks = (int)((sthreads + 255) / 256);

    // ---- layer 1 ----
    scatter_accum_kernel<<<sblocks, 256, 0, stream>>>(xu, xm, esrc, edst,
                                                      agg_m, agg_u, E);
    sage_gemm_kernel<true><<<(NM + 15) / 16, 128, 0, stream>>>(
        agg_m, deg_m, xm, wfrag + 0 * 8192, wfrag + 1 * 8192, b1_um, r_m, NM);
    sage_gemm_kernel<true><<<(NU + 15) / 16, 128, 0, stream>>>(
        agg_u, deg_u, xu, wfrag + 2 * 8192, wfrag + 3 * 8192, b1_mu, r_u, NU);

    // ---- layer 2 ----
    zero_f32_kernel<<<zgrid, zblk, 0, stream>>>(agg_u, zero_agg_n);
    scatter_accum_kernel<<<sblocks, 256, 0, stream>>>(r_u, r_m, esrc, edst,
                                                      agg_m, agg_u, E);
    sage_gemm_kernel<false><<<(NM + 15) / 16, 128, 0, stream>>>(
        agg_m, deg_m, r_m, wfrag + 4 * 8192, wfrag + 5 * 8192, b2_um, out_m, NM);
    sage_gemm_kernel<false><<<(NU + 15) / 16, 128, 0, stream>>>(
        agg_u, deg_u, r_u, wfrag + 6 * 8192, wfrag + 7 * 8192, b2_mu, out_u, NU);
}
